// NoExtraEvoformerIteration_7232724927145
// MI455X (gfx1250) — matrix operand order repaired and checked by reading, not compiled, not run on hardware
//
#include <hip/hip_runtime.h>

typedef __attribute__((ext_vector_type(16))) _Float16 v16h;
typedef __attribute__((ext_vector_type(8)))  _Float16 v8h;
typedef __attribute__((ext_vector_type(8)))  float    v8f;
typedef __attribute__((ext_vector_type(4)))  float    vf4;
typedef long long i64;

// global (addrspace 1) pointer views -> global_load_* instead of flat_load_*
typedef const __attribute__((address_space(1))) float* gfp;
typedef const __attribute__((address_space(1))) vf4*   gf4p;

#define GT_M 64
#define GT_N 64
#define GT_K 32
#define LPAD 8

// ---------------------------------------------------------------------------
// Generic strided/batched WMMA GEMM:
//   C[z][m][n] = act(alpha * sum_k A[z][m][k] * B[z][n][k]  + bias[n])
// A element addr = aoff + z0*az0 + z1*az1 + m*ams + k*aks   (z0=z%Z0, z1=z/Z0)
// B element addr = boff + z0*bz0 + z1*bz1 + n*bns + k*bks
// C element addr (smode 0) = coff + z0*cz0 + z1*cz1 + m*cms + n*cns
//                (smode 1) = coff + z0*cz0 + (n>>5)*1024 + m*32 + (n&31)  [OPM]
// K multiple of 32. f32->f16 in LDS, f32 accum via v_wmma_f32_16x16x32_f16.
// Block = 256 threads = 8 wave32. Tile 64x64: wave (wm,wn) owns m-subtiles
// wm*16 and 32+wm*16 at n-subtile wn*16 -> 2 WMMAs/K-step sharing one B frag.
// Staging: thread owns 8 k-consecutive elements (rr=tid>>2, kk0=(tid&3)*8);
// per-lane offset is loop-invariant, only the base advances per K-step.
// AVEC/BVEC: k-stride==1 -> clause-paired global_load_b128.
// ---------------------------------------------------------------------------
template <bool AVEC, bool BVEC>
__global__ __launch_bounds__(256) void k_gemm_t(
    const float* __restrict__ A, const float* __restrict__ B,
    const float* __restrict__ bias, float* __restrict__ C,
    int M, int N, int K,
    i64 aoff, i64 ams, i64 aks, i64 az0, i64 az1, int Z0,
    i64 boff, i64 bns, i64 bks, i64 bz0, i64 bz1,
    i64 coff, i64 cms, i64 cns, i64 cz0, i64 cz1,
    float alpha, int act, int smode)
{
  __shared__ _Float16 As[GT_M][GT_K + LPAD];
  __shared__ _Float16 Bs[GT_N][GT_K + LPAD];
  const int tid = threadIdx.x;
  const int z   = blockIdx.z;
  const int z0  = z % Z0;
  const int z1  = z / Z0;
  const i64 ab = aoff + (i64)z0 * az0 + (i64)z1 * az1;
  const i64 bb = boff + (i64)z0 * bz0 + (i64)z1 * bz1;
  const i64 cb = coff + (i64)z0 * cz0 + (i64)z1 * cz1;
  const int m0 = blockIdx.y * GT_M;
  const int n0 = blockIdx.x * GT_N;
  const int lane = tid & 31;
  const int wave = tid >> 5;
  const int wm = wave & 1;       // m-subtiles wm*16 and 32+wm*16
  const int wn = wave >> 1;      // n-subtile wn*16
  const int l16 = lane & 15;
  const int lh  = lane >> 4;

  // staging coords: row rr, k-chunk kk0..kk0+7 (covers 64x32 with 256 thr)
  const int rr  = tid >> 2;
  const int kk0 = (tid & 3) * 8;
  int gm = m0 + rr;
  const bool okA = gm < M; if (!okA) gm = m0;      // clamp to valid row
  int gn = n0 + rr;
  const bool okB = gn < N; if (!okB) gn = n0;
  const i64 oA = (i64)(gm - m0) * ams + (i64)kk0 * aks;  // within-slab offset
  const i64 oB = (i64)(gn - n0) * bns + (i64)kk0 * bks;

  // global bases (advance per K-step); per-lane offset stays constant
  gfp Ag = (gfp)(unsigned long long)(A + ab + (i64)m0 * ams);
  gfp Bg = (gfp)(unsigned long long)(B + bb + (i64)n0 * bns);
  const i64 astep = (i64)GT_K * aks;
  const i64 bstep = (i64)GT_K * bks;

  v8f acc0 = {}, acc1 = {};
  for (int k0 = 0; k0 < K; k0 += GT_K) {
    float va[8], vb[8];
    if (AVEC) {                       // k-contiguous: two 16B loads
      vf4 a0 = *(gf4p)(Ag + oA);
      vf4 a1 = *(gf4p)(Ag + oA + 4);
      #pragma unroll
      for (int t = 0; t < 4; ++t) { va[t] = a0[t]; va[t + 4] = a1[t]; }
    } else {
      #pragma unroll
      for (int t = 0; t < 8; ++t) va[t] = Ag[oA + (i64)t * aks];
    }
    if (BVEC) {
      vf4 b0 = *(gf4p)(Bg + oB);
      vf4 b1 = *(gf4p)(Bg + oB + 4);
      #pragma unroll
      for (int t = 0; t < 4; ++t) { vb[t] = b0[t]; vb[t + 4] = b1[t]; }
    } else {
      #pragma unroll
      for (int t = 0; t < 8; ++t) vb[t] = Bg[oB + (i64)t * bks];
    }
    if (k0 + GT_K < K) {   // gfx1250 global_prefetch_b8 of next K slab
      __builtin_prefetch((const float*)(Ag + oA) + astep, 0, 1);
      __builtin_prefetch((const float*)(Bg + oB) + bstep, 0, 1);
    }
    __syncthreads();       // previous iteration's fragment reads complete
    v8h ha, hb;
    #pragma unroll
    for (int t = 0; t < 8; ++t) {
      ha[t] = okA ? (_Float16)va[t] : (_Float16)0.0f;
      hb[t] = okB ? (_Float16)vb[t] : (_Float16)0.0f;
    }
    *(v8h*)&As[rr][kk0] = ha;         // one ds_store_b128 each
    *(v8h*)&Bs[rr][kk0] = hb;
    __syncthreads();
    // A fragment 16x32 (ISA 7.12.2): lane-half selects K {h*8..+7, 16+h*8..+7}
    v16h af0, af1, bf;
    #pragma unroll
    for (int e = 0; e < 8; ++e) {
      af0[e]     = As[wm * 16 + l16]     [lh * 8 + e];
      af0[e + 8] = As[wm * 16 + l16]     [16 + lh * 8 + e];
      af1[e]     = As[32 + wm * 16 + l16][lh * 8 + e];
      af1[e + 8] = As[32 + wm * 16 + l16][16 + lh * 8 + e];
    }
    // B fragment 32x16: lane-half selects K {h*8..+7, 16+h*8..+7} at column l16
    #pragma unroll
    for (int e = 0; e < 16; ++e) bf[e] = Bs[wn * 16 + l16][(e < 8 ? lh * 8 + e : 16 + lh * 8 + (e - 8))];
    acc0 = __builtin_amdgcn_wmma_f32_16x16x32_f16(false, af0, false, bf,
                                                  (short)0, acc0, false, false);
    acc1 = __builtin_amdgcn_wmma_f32_16x16x32_f16(false, af1, false, bf,
                                                  (short)0, acc1, false, false);
    Ag += astep;
    Bg += bstep;
  }
  // C/D layout: lane = col(l16) + 16*rowhalf; vgpr r -> row r + 8*rowhalf
  const int n = n0 + wn * 16 + l16;
  if (n < N) {
    const float bv = bias ? bias[n] : 0.0f;
    #pragma unroll
    for (int r = 0; r < 8; ++r) {
      const int ma = m0 + wm * 16 + lh * 8 + r;
      const int mb = ma + 32;
      float v0 = alpha * acc0[r] + bv;
      float v1 = alpha * acc1[r] + bv;
      if (act == 1) { v0 = fmaxf(v0, 0.0f); v1 = fmaxf(v1, 0.0f); }
      else if (act == 2) {
        v0 = 1.0f / (1.0f + __expf(-v0));
        v1 = 1.0f / (1.0f + __expf(-v1));
      }
      if (ma < M) {
        i64 a0 = (smode == 0)
               ? cb + (i64)ma * cms + (i64)n * cns
               : cb + (i64)(n >> 5) * 1024 + (i64)ma * 32 + (i64)(n & 31);
        C[a0] = v0;
      }
      if (mb < M) {
        i64 a1 = (smode == 0)
               ? cb + (i64)mb * cms + (i64)n * cns
               : cb + (i64)(n >> 5) * 1024 + (i64)mb * 32 + (i64)(n & 31);
        C[a1] = v1;
      }
    }
  }
}

// ---------------------------------------------------------------------------
// LayerNorm over last dim C (one block of 128 threads per row)
// ---------------------------------------------------------------------------
__global__ __launch_bounds__(128) void k_ln(
    const float* __restrict__ x, const float* __restrict__ sc,
    const float* __restrict__ bi, float* __restrict__ y, int C)
{
  const i64 row = blockIdx.x;
  const float* xr = x + row * (i64)C;
  float* yr = y + row * (i64)C;
  __shared__ float red[128];
  float s = 0.0f;
  for (int c = threadIdx.x; c < C; c += 128) s += xr[c];
  red[threadIdx.x] = s; __syncthreads();
  for (int o = 64; o > 0; o >>= 1) {
    if (threadIdx.x < o) red[threadIdx.x] += red[threadIdx.x + o];
    __syncthreads();
  }
  const float mean = red[0] / (float)C; __syncthreads();
  float v = 0.0f;
  for (int c = threadIdx.x; c < C; c += 128) { float d = xr[c] - mean; v += d * d; }
  red[threadIdx.x] = v; __syncthreads();
  for (int o = 64; o > 0; o >>= 1) {
    if (threadIdx.x < o) red[threadIdx.x] += red[threadIdx.x + o];
    __syncthreads();
  }
  const float rstd = rsqrtf(red[0] / (float)C + 1e-5f);
  for (int c = threadIdx.x; c < C; c += 128)
    yr[c] = (xr[c] - mean) * rstd * sc[c] + bi[c];
}

// ---------------------------------------------------------------------------
// Softmax over K (<=256) with mask bias 1e9*(mask[b,k]-1) and optional
// nb[h,q,k]. One block (128 thr) per logits row; rows ordered [b][h][q].
// gfx1250 async path: the row is staged into LDS once with
// GLOBAL_LOAD_ASYNC_TO_LDS_B32 (+ s_wait_asynccnt), then the max/exp/sum
// passes run out of LDS; only the final normalized row is written to HBM.
// ---------------------------------------------------------------------------
__global__ __launch_bounds__(128) void k_softmax(
    float* __restrict__ lg, const float* __restrict__ mask,
    const float* __restrict__ nb, int H, int Q, int K)
{
  __shared__ float srow[256];
  __shared__ float red[128];
  const i64 r = blockIdx.x;
  const int q = (int)(r % Q);
  const int h = (int)((r / Q) % H);
  const i64 b = r / ((i64)Q * H);
  float* row = lg + r * (i64)K;
  const float* mrow = mask + b * (i64)K;
  const float* nbrow = nb ? nb + ((i64)h * Q + q) * (i64)K : (const float*)0;

  // async-stage logits row into LDS (ASYNCcnt path)
  for (int k = threadIdx.x; k < K; k += 128) {
    unsigned ldsoff = (unsigned)(unsigned long long)(&srow[k]);
    unsigned long long ga = (unsigned long long)(row + k);
    asm volatile("global_load_async_to_lds_b32 %0, %1, off"
                 :: "v"(ldsoff), "v"(ga) : "memory");
  }
  asm volatile("s_wait_asynccnt 0" ::: "memory");
  __syncthreads();

  float mx = -3.0e38f;
  for (int k = threadIdx.x; k < K; k += 128) {
    float v = srow[k] + 1e9f * (mrow[k] - 1.0f);
    if (nbrow) v += nbrow[k];
    srow[k] = v;
    mx = fmaxf(mx, v);
  }
  red[threadIdx.x] = mx; __syncthreads();
  for (int o = 64; o > 0; o >>= 1) {
    if (threadIdx.x < o) red[threadIdx.x] = fmaxf(red[threadIdx.x], red[threadIdx.x + o]);
    __syncthreads();
  }
  mx = red[0]; __syncthreads();
  float s = 0.0f;
  for (int k = threadIdx.x; k < K; k += 128) {
    float e = __expf(srow[k] - mx); srow[k] = e; s += e;
  }
  red[threadIdx.x] = s; __syncthreads();
  for (int o = 64; o > 0; o >>= 1) {
    if (threadIdx.x < o) red[threadIdx.x] += red[threadIdx.x + o];
    __syncthreads();
  }
  const float inv = 1.0f / red[0];
  for (int k = threadIdx.x; k < K; k += 128) row[k] = srow[k] * inv;
}

// ------------------------- elementwise helpers ------------------------------
__global__ void k_add_out(float* __restrict__ d, const float* __restrict__ a,
                          const float* __restrict__ b, i64 n) {
  for (i64 i = blockIdx.x * (i64)blockDim.x + threadIdx.x; i < n;
       i += (i64)gridDim.x * blockDim.x) d[i] = a[i] + b[i];
}
__global__ void k_add_inplace(float* __restrict__ d, const float* __restrict__ a, i64 n) {
  for (i64 i = blockIdx.x * (i64)blockDim.x + threadIdx.x; i < n;
       i += (i64)gridDim.x * blockDim.x) d[i] += a[i];
}
__global__ void k_mul_inplace(float* __restrict__ d, const float* __restrict__ a, i64 n) {
  for (i64 i = blockIdx.x * (i64)blockDim.x + threadIdx.x; i < n;
       i += (i64)gridDim.x * blockDim.x) d[i] *= a[i];
}
__global__ void k_mul_add(float* __restrict__ d, const float* __restrict__ a,
                          const float* __restrict__ b, i64 n) {   // d += a*b
  for (i64 i = blockIdx.x * (i64)blockDim.x + threadIdx.x; i < n;
       i += (i64)gridDim.x * blockDim.x) d[i] += a[i] * b[i];
}
__global__ void k_mask_rows(float* __restrict__ x, const float* __restrict__ m,
                            i64 n, int C) {                       // x[r,c]*=m[r]
  for (i64 i = blockIdx.x * (i64)blockDim.x + threadIdx.x; i < n;
       i += (i64)gridDim.x * blockDim.x) x[i] *= m[i / C];
}
__global__ void k_opm_finish(float* __restrict__ pr, const float* __restrict__ pin,
                             const float* __restrict__ acc,
                             const float* __restrict__ nrm, i64 n, int C) {
  for (i64 i = blockIdx.x * (i64)blockDim.x + threadIdx.x; i < n;
       i += (i64)gridDim.x * blockDim.x)
    pr[i] = pin[i] + acc[i] / (1e-3f + nrm[i / C]);
}
__global__ void k_transpose01(float* __restrict__ dst, const float* __restrict__ src,
                              int D0, int D1, int C) {            // [D0,D1,C]->[D1,D0,C]
  const i64 n = (i64)D0 * D1 * C;
  for (i64 i = blockIdx.x * (i64)blockDim.x + threadIdx.x; i < n;
       i += (i64)gridDim.x * blockDim.x) {
    i64 c = i % C; i64 t = i / C; i64 d1 = t % D1; i64 d0 = t / D1;
    dst[(d1 * D0 + d0) * C + c] = src[i];
  }
}

// ---------------------------------------------------------------------------
// Host orchestration
// ---------------------------------------------------------------------------

// Param leaf order: jax pytree flatten (sorted dict keys) after the 4 tensors.
enum {
  IN_MSA = 0, IN_MSA_MASK, IN_PAIR, IN_PAIR_MASK,
  CA_GATING_B = 4, CA_GATING_W, CA_K_W, CA_OUT_B, CA_OUT_W, CA_Q_W, CA_V_W,
  CA_QN_BIAS, CA_QN_SCALE,                                       // 11,12
  MT_L1_B = 13, MT_L1_W, MT_L2_B, MT_L2_W, MT_N_BIAS, MT_N_SCALE, // ..18
  OPM_L_B = 19, OPM_L_W, OPM_N_BIAS, OPM_N_SCALE, OPM_OUT_B, OPM_OUT_W,
  OPM_R_B, OPM_R_W,                                              // ..26
  PT_L1_B = 27, PT_L1_W, PT_L2_B, PT_L2_W, PT_N_BIAS, PT_N_SCALE, // ..32
  RA_GATING_B = 33, RA_GATING_W, RA_K_W, RA_OUT_B, RA_OUT_W, RA_Q_W, RA_V_W,
  RA_2DN_BIAS = 40, RA_2DN_SCALE, RA_2DW, RA_QN_BIAS, RA_QN_SCALE, // ..44
  TAE_GATING_B = 45, TAE_GATING_W, TAE_K_W, TAE_OUT_B, TAE_OUT_W, TAE_Q_W, TAE_V_W,
  TAE_2DW = 52, TAE_QN_BIAS, TAE_QN_SCALE,                       // ..54
  TAS_GATING_B = 55, TAS_GATING_W, TAS_K_W, TAS_OUT_B, TAS_OUT_W, TAS_Q_W, TAS_V_W,
  TAS_2DW = 62, TAS_QN_BIAS, TAS_QN_SCALE,                       // ..64
  TI_GATE_B = 65, TI_GATE_W, TI_L_B, TI_L_W, TI_LG_B, TI_LG_W,
  TI_NC_BIAS, TI_NC_SCALE, TI_NI_BIAS, TI_NI_SCALE,
  TI_OUT_B, TI_OUT_W, TI_RG_B, TI_RG_W, TI_R_B, TI_R_W,          // ..80
  TO_GATE_B = 81, TO_GATE_W, TO_L_B, TO_L_W, TO_LG_B, TO_LG_W,
  TO_NC_BIAS, TO_NC_SCALE, TO_NI_BIAS, TO_NI_SCALE,
  TO_OUT_B, TO_OUT_W, TO_RG_B, TO_RG_W, TO_R_B, TO_R_W           // ..96
};

extern "C" void kernel_launch(void* const* d_in, const int* in_sizes, int n_in,
                              void* d_out, int out_size, void* d_ws, size_t ws_size,
                              hipStream_t stream)
{
  (void)in_sizes; (void)n_in; (void)out_size; (void)ws_size;
  const int S = 128, R = 256, CM = 256, CZ = 128, HM = 8, HT = 4;
  const i64 MSA_N  = (i64)S * R * CM;    // 8388608
  const i64 PAIR_N = (i64)R * R * CZ;    // 8388608
  const i64 MR  = (i64)S * R;            // 32768 msa rows
  const i64 PRr = (i64)R * R;            // 65536 pair rows
  const float QS = 0.17677669529663687f; // 1/sqrt(32)

  auto P = [&](int i) { return (const float*)d_in[i]; };
  const float* msa_in    = P(IN_MSA);
  const float* msa_mask  = P(IN_MSA_MASK);
  const float* pair_in   = P(IN_PAIR);
  const float* pair_mask = P(IN_PAIR_MASK);

  float* msa  = (float*)d_out;            // output msa written fully by row-att
  float* pair = (float*)d_out + MSA_N;    // output pair written fully by OPM

  // bump allocator on workspace
  size_t off = 0; char* wsb = (char*)d_ws;
  auto alloc = [&](i64 nf) -> float* {
    float* p = (float*)(wsb + off);
    off += ((size_t)nf * 4 + 255) & ~(size_t)255;
    return p;
  };
  float* B0  = alloc(MSA_N);  // LN(act)
  float* B1  = alloc(MSA_N);  // q
  float* B2  = alloc(MSA_N);  // k
  float* B3  = alloc(MSA_N);  // v
  float* B4  = alloc(MSA_N);  // gates
  float* B5  = alloc(MSA_N);  // wv
  float* B6  = alloc(MSA_N);  // module out
  float* B7  = alloc(MSA_N);  // transposed act / triangle product
  float* B8  = alloc(MSA_N);  // z_ln / tri left
  float* B9  = alloc(MSA_N);  // tri right
  float* B10 = alloc(MSA_N);  // OPM acc / transpose temp
  float* B11 = alloc(MSA_N);  // tri norm_c LN
  float* LOG = alloc((i64)S * HM * R * R);  // 67.1M f: logits / FFN hidden
  float* T1  = alloc((i64)16 * R * 1024);   // OPM chunk scratch
  float* NB  = alloc((i64)HM * R * R);      // nonbatched bias
  float* NRM = alloc(PRr);
  float* MKT = alloc(PRr);                  // transposed masks

  const dim3 EW(4096), TB(256);

  auto gemm = [&](const float* Ap, const float* Bp, const float* bp, float* Cp,
                  int M, int N, int K, int Z, int Z0,
                  i64 ao, i64 ams, i64 aks, i64 az0, i64 az1,
                  i64 bo, i64 bns, i64 bks, i64 bz0, i64 bz1,
                  i64 co, i64 cms, i64 cns, i64 cz0, i64 cz1,
                  float alpha, int act, int smode) {
    dim3 g((N + GT_N - 1) / GT_N, (M + GT_M - 1) / GT_M, Z);
    if (aks == 1 && bks == 1)
      k_gemm_t<true, true><<<g, TB, 0, stream>>>(Ap, Bp, bp, Cp, M, N, K,
          ao, ams, aks, az0, az1, Z0, bo, bns, bks, bz0, bz1,
          co, cms, cns, cz0, cz1, alpha, act, smode);
    else if (aks == 1)
      k_gemm_t<true, false><<<g, TB, 0, stream>>>(Ap, Bp, bp, Cp, M, N, K,
          ao, ams, aks, az0, az1, Z0, bo, bns, bks, bz0, bz1,
          co, cms, cns, cz0, cz1, alpha, act, smode);
    else if (bks == 1)
      k_gemm_t<false, true><<<g, TB, 0, stream>>>(Ap, Bp, bp, Cp, M, N, K,
          ao, ams, aks, az0, az1, Z0, bo, bns, bks, bz0, bz1,
          co, cms, cns, cz0, cz1, alpha, act, smode);
    else
      k_gemm_t<false, false><<<g, TB, 0, stream>>>(Ap, Bp, bp, Cp, M, N, K,
          ao, ams, aks, az0, az1, Z0, bo, bns, bks, bz0, bz1,
          co, cms, cns, cz0, cz1, alpha, act, smode);
  };
  // plain row-major: C[M,N] = act(alpha*A[M,K] @ W[K,N] + b)
  auto gemm_nn = [&](const float* Ap, const float* Wp, const float* bp, float* Cp,
                     int M, int N, int K, float alpha, int act) {
    gemm(Ap, Wp, bp, Cp, M, N, K, 1, 1,
         0, K, 1, 0, 0,  0, 1, N, 0, 0,  0, N, 1, 0, 0, alpha, act, 0);
  };
  auto ln = [&](const float* x, float* y, const float* sc, const float* bi,
                i64 rows, int C) {
    k_ln<<<dim3((unsigned)rows), dim3(128), 0, stream>>>(x, sc, bi, y, C);
  };
  auto softmax = [&](float* lg, const float* mk, const float* nb,
                     i64 rows, int H, int Q, int K) {
    k_softmax<<<dim3((unsigned)rows), dim3(128), 0, stream>>>(lg, mk, nb, H, Q, K);
  };

  // Full gating-attention on pre-LN'ed act (q_data == m_data == actln).
  // rows = B*Q, C = H*32, d = 32.
  auto attention = [&](const float* actln, i64 rows, int C, int H,
                       int Bb, int Q, int Kk,
                       const float* qw, const float* kw, const float* vw,
                       const float* gw, const float* gb,
                       const float* ow, const float* ob,
                       const float* maskp, const float* nbp, float* outb) {
    const int HD = C;
    gemm_nn(actln, qw, nullptr, B1, (int)rows, HD, C, QS,   0);
    gemm_nn(actln, kw, nullptr, B2, (int)rows, HD, C, 1.0f, 0);
    gemm_nn(actln, vw, nullptr, B3, (int)rows, HD, C, 1.0f, 0);
    gemm_nn(actln, gw, gb,      B4, (int)rows, HD, C, 1.0f, 2);  // sigmoid gate
    const i64 zqk = (i64)Q * Kk;        // per (b,h) logits stride
    const i64 zab = (i64)Q * HD;        // per b stride in q/k/v (Q==keys rows)
    // logits[b,h,q,k] = q . k
    gemm(B1, B2, nullptr, LOG, Q, Kk, 32, Bb * H, H,
         0, HD, 1, 32, zab,   0, HD, 1, 32, zab,
         0, Kk, 1, zqk, (i64)H * zqk, 1.0f, 0, 0);
    softmax(LOG, maskp, nbp, (i64)Bb * H * Q, H, Q, Kk);
    // wv[b,q,h,d] = w . v
    gemm(LOG, B3, nullptr, B5, Q, 32, Kk, Bb * H, H,
         0, Kk, 1, zqk, (i64)H * zqk,   0, 1, HD, 32, zab,
         0, HD, 1, 32, zab, 1.0f, 0, 0);
    k_mul_inplace<<<EW, TB, 0, stream>>>(B5, B4, rows * (i64)HD);
    gemm_nn(B5, ow, ob, outb, (int)rows, C, HD, 1.0f, 0);
  };

  // ======================= 1. MSA row attention ============================
  ln(msa_in,  B0, P(RA_QN_SCALE),  P(RA_QN_BIAS),  MR,  CM);
  ln(pair_in, B8, P(RA_2DN_SCALE), P(RA_2DN_BIAS), PRr, CZ);
  // nb[h][q*R+k] = sum_c z[q,k,c] * W2d[c,h]
  gemm(B8, P(RA_2DW), nullptr, NB, (int)PRr, HM, CZ, 1, 1,
       0, CZ, 1, 0, 0,  0, 1, HM, 0, 0,  0, 1, PRr, 0, 0, 1.0f, 0, 0);
  attention(B0, MR, CM, HM, S, R, R,
            P(RA_Q_W), P(RA_K_W), P(RA_V_W), P(RA_GATING_W), P(RA_GATING_B),
            P(RA_OUT_W), P(RA_OUT_B), msa_mask, NB, B6);
  k_add_out<<<EW, TB, 0, stream>>>(msa, msa_in, B6, MSA_N);

  // ======================= 2. MSA column attention =========================
  k_transpose01<<<EW, TB, 0, stream>>>(B7, msa, S, R, CM);     // [R,S,C]
  k_transpose01<<<EW, TB, 0, stream>>>(MKT, msa_mask, S, R, 1); // [R,S]
  ln(B7, B0, P(CA_QN_SCALE), P(CA_QN_BIAS), MR, CM);
  attention(B0, MR, CM, HM, R, S, S,
            P(CA_Q_W), P(CA_K_W), P(CA_V_W), P(CA_GATING_W), P(CA_GATING_B),
            P(CA_OUT_W), P(CA_OUT_B), MKT, nullptr, B6);
  k_transpose01<<<EW, TB, 0, stream>>>(B7, B6, R, S, CM);      // back to [S,R,C]
  k_add_inplace<<<EW, TB, 0, stream>>>(msa, B7, MSA_N);

  // ======================= 3. MSA transition ===============================
  ln(msa, B0, P(MT_N_SCALE), P(MT_N_BIAS), MR, CM);
  gemm_nn(B0,  P(MT_L1_W), P(MT_L1_B), LOG, (int)MR, 4 * CM, CM,     1.0f, 1);
  gemm_nn(LOG, P(MT_L2_W), P(MT_L2_B), B6,  (int)MR, CM,     4 * CM, 1.0f, 0);
  k_add_inplace<<<EW, TB, 0, stream>>>(msa, B6, MSA_N);

  // ======================= 4. Outer product mean ===========================
  ln(msa, B0, P(OPM_N_SCALE), P(OPM_N_BIAS), MR, CM);
  gemm_nn(B0, P(OPM_L_W), P(OPM_L_B), B8, (int)MR, 32, CM, 1.0f, 0);
  k_mask_rows<<<EW, TB, 0, stream>>>(B8, msa_mask, MR * 32, 32);
  gemm_nn(B0, P(OPM_R_W), P(OPM_R_B), B9, (int)MR, 32, CM, 1.0f, 0);
  k_mask_rows<<<EW, TB, 0, stream>>>(B9, msa_mask, MR * 32, 32);
  // norm[i,j] = sum_s mask[s,i]*mask[s,j]
  gemm(msa_mask, msa_mask, nullptr, NRM, R, R, S, 1, 1,
       0, 1, R, 0, 0,  0, 1, R, 0, 0,  0, R, 1, 0, 0, 1.0f, 0, 0);
  // outer: chunk 16 i's; T1[z][j][c*32+e] = sum_s L[s,i,c] R[s,j,e]
  for (int ic = 0; ic < 16; ++ic) {
    gemm(B8, B9, nullptr, T1, 32, 8192, S, 16, 16,
         (i64)ic * 512, 1, 8192, 32, 0,
         0, 1, 8192, 0, 0,
         0, 0, 0, 262144, 0, 1.0f, 0, /*opm swizzle*/1);
    // out[i,j,f] = T1[z][j][:] @ W[1024,128] + b
    gemm(T1, P(OPM_OUT_W), P(OPM_OUT_B), B10, R, CZ, 1024, 16, 16,
         0, 1024, 1, 262144, 0,
         0, 1, CZ, 0, 0,
         (i64)ic * 524288, CZ, 1, (i64)R * CZ, 0, 1.0f, 0, 0);
  }
  k_opm_finish<<<EW, TB, 0, stream>>>(pair, pair_in, B10, NRM, PAIR_N, CZ);

  // ======================= 5/6. Triangle multiplications ===================
  auto trimul = [&](const float* gb_, const float* gw_, const float* lb_,
                    const float* lw_, const float* lgb_, const float* lgw_,
                    const float* ncb_, const float* ncs_, const float* nib_,
                    const float* nis_, const float* ob_, const float* ow_,
                    const float* rgb_, const float* rgw_, const float* rb_,
                    const float* rw_, bool outgoing) {
    ln(pair, B0, nis_, nib_, PRr, CZ);
    gemm_nn(B0, lw_,  lb_,  B8, (int)PRr, CZ, CZ, 1.0f, 0);
    gemm_nn(B0, lgw_, lgb_, B4, (int)PRr, CZ, CZ, 1.0f, 2);
    k_mul_inplace<<<EW, TB, 0, stream>>>(B8, B4, PAIR_N);
    k_mask_rows<<<EW, TB, 0, stream>>>(B8, pair_mask, PAIR_N, CZ);
    gemm_nn(B0, rw_,  rb_,  B9, (int)PRr, CZ, CZ, 1.0f, 0);
    gemm_nn(B0, rgw_, rgb_, B4, (int)PRr, CZ, CZ, 1.0f, 2);
    k_mul_inplace<<<EW, TB, 0, stream>>>(B9, B4, PAIR_N);
    k_mask_rows<<<EW, TB, 0, stream>>>(B9, pair_mask, PAIR_N, CZ);
    const i64 RS = (i64)R * CZ;  // 32768
    if (outgoing)  // x[i,j,c] = sum_k L[i,k,c]*R[j,k,c]
      gemm(B8, B9, nullptr, B7, R, R, R, CZ, CZ,
           0, RS, CZ, 1, 0,  0, RS, CZ, 1, 0,  0, RS, CZ, 1, 0, 1.0f, 0, 0);
    else           // x[i,j,c] = sum_k R[k,i,c]*L[k,j,c]
      gemm(B9, B8, nullptr, B7, R, R, R, CZ, CZ,
           0, CZ, RS, 1, 0,  0, CZ, RS, 1, 0,  0, RS, CZ, 1, 0, 1.0f, 0, 0);
    ln(B7, B11, ncs_, ncb_, PRr, CZ);
    gemm_nn(B11, ow_, ob_, B6, (int)PRr, CZ, CZ, 1.0f, 0);
    gemm_nn(B0,  gw_, gb_, B4, (int)PRr, CZ, CZ, 1.0f, 2);
    k_mul_add<<<EW, TB, 0, stream>>>(pair, B6, B4, PAIR_N);  // pair += x*g
  };
  trimul(P(TO_GATE_B), P(TO_GATE_W), P(TO_L_B), P(TO_L_W), P(TO_LG_B), P(TO_LG_W),
         P(TO_NC_BIAS), P(TO_NC_SCALE), P(TO_NI_BIAS), P(TO_NI_SCALE),
         P(TO_OUT_B), P(TO_OUT_W), P(TO_RG_B), P(TO_RG_W), P(TO_R_B), P(TO_R_W), true);
  trimul(P(TI_GATE_B), P(TI_GATE_W), P(TI_L_B), P(TI_L_W), P(TI_LG_B), P(TI_LG_W),
         P(TI_NC_BIAS), P(TI_NC_SCALE), P(TI_NI_BIAS), P(TI_NI_SCALE),
         P(TI_OUT_B), P(TI_OUT_W), P(TI_RG_B), P(TI_RG_W), P(TI_R_B), P(TI_R_W), false);

  // ======================= 7. Triangle attention (start) ===================
  ln(pair, B0, P(TAS_QN_SCALE), P(TAS_QN_BIAS), PRr, CZ);
  gemm(B0, P(TAS_2DW), nullptr, NB, (int)PRr, HT, CZ, 1, 1,
       0, CZ, 1, 0, 0,  0, 1, HT, 0, 0,  0, 1, PRr, 0, 0, 1.0f, 0, 0);
  attention(B0, PRr, CZ, HT, R, R, R,
            P(TAS_Q_W), P(TAS_K_W), P(TAS_V_W), P(TAS_GATING_W), P(TAS_GATING_B),
            P(TAS_OUT_W), P(TAS_OUT_B), pair_mask, NB, B6);
  k_add_inplace<<<EW, TB, 0, stream>>>(pair, B6, PAIR_N);

  // ======================= 8. Triangle attention (end) =====================
  k_transpose01<<<EW, TB, 0, stream>>>(B7, pair, R, R, CZ);
  k_transpose01<<<EW, TB, 0, stream>>>(MKT, pair_mask, R, R, 1);
  ln(B7, B0, P(TAE_QN_SCALE), P(TAE_QN_BIAS), PRr, CZ);
  gemm(B0, P(TAE_2DW), nullptr, NB, (int)PRr, HT, CZ, 1, 1,
       0, CZ, 1, 0, 0,  0, 1, HT, 0, 0,  0, 1, PRr, 0, 0, 1.0f, 0, 0);
  attention(B0, PRr, CZ, HT, R, R, R,
            P(TAE_Q_W), P(TAE_K_W), P(TAE_V_W), P(TAE_GATING_W), P(TAE_GATING_B),
            P(TAE_OUT_W), P(TAE_OUT_B), MKT, NB, B6);
  k_transpose01<<<EW, TB, 0, stream>>>(B10, B6, R, R, CZ);
  k_add_inplace<<<EW, TB, 0, stream>>>(pair, B10, PAIR_N);

  // ======================= 9. Pair transition ==============================
  ln(pair, B0, P(PT_N_SCALE), P(PT_N_BIAS), PRr, CZ);
  gemm_nn(B0,  P(PT_L1_W), P(PT_L1_B), LOG, (int)PRr, 4 * CZ, CZ,     1.0f, 1);
  gemm_nn(LOG, P(PT_L2_W), P(PT_L2_B), B6,  (int)PRr, CZ,     4 * CZ, 1.0f, 0);
  k_add_inplace<<<EW, TB, 0, stream>>>(pair, B6, PAIR_N);
}
